// HyFreqMoE_SR4_4801773437221
// MI455X (gfx1250) — matrix operand order repaired and checked by reading, not compiled, not run on hardware
//
#include <hip/hip_runtime.h>
#include <math.h>

// ---------------------------------------------------------------------------
// Types for CDNA5 WMMA (wave32): v16bf A/B fragments, v8f accumulator.
// ---------------------------------------------------------------------------
typedef __bf16 bf16;
typedef __bf16  v16bf __attribute__((ext_vector_type(16)));
typedef __bf16  v8bf  __attribute__((ext_vector_type(8)));
typedef float   v8f   __attribute__((ext_vector_type(8)));
typedef unsigned int u32x4 __attribute__((ext_vector_type(4)));
typedef int          i32x4 __attribute__((ext_vector_type(4)));
typedef int          i32x8 __attribute__((ext_vector_type(8)));

#define HH 192
#define WWID 192
#define HWC (HH * WWID)     // 36864
#define BATCH 2
#define CCH 128
#define WSZ 8
#define SSH 4
#define TT 64               // WSZ*WSZ
#define NWIN 576            // (192/8)^2

#if defined(__gfx1250__) && __has_builtin(__builtin_amdgcn_tensor_load_to_lds)
#define HAVE_TDM 1
#else
#define HAVE_TDM 0
#endif

#if defined(__gfx1250__) && __has_builtin(__builtin_amdgcn_s_wait_tensorcnt)
#define WAIT_TENSOR() __builtin_amdgcn_s_wait_tensorcnt(0)
#else
#define WAIT_TENSOR()
#endif

static __device__ __forceinline__ v8f wmma_bf16(const v16bf& a, const v16bf& b, const v8f& c) {
    return __builtin_amdgcn_wmma_f32_16x16x32_bf16(false, a, false, b, (short)0, c, false, false);
}

static __device__ __forceinline__ v16bf cat8(v8bf lo, v8bf hi) {
    return __builtin_shufflevector(lo, hi, 0, 1, 2, 3, 4, 5, 6, 7,
                                   8, 9, 10, 11, 12, 13, 14, 15);
}

#if HAVE_TDM
// ---------------------------------------------------------------------------
// Tensor Data Mover: 2D bf16 tile (tile_w x tile_h elements, row stride in
// elements) from global memory into LDS at byte offset lds_off.  D# bitfields
// per CDNA5 ISA 8.3/8.4: group0 = {count/type, lds_addr, global_addr},
// group1 = {data_size, tensor dims, tile dims, dim0 stride}.  This toolchain
// exposes the 6-arg builtin (g0, g1, g2, g3, extra group, cpol).
// ---------------------------------------------------------------------------
static __device__ __forceinline__ void tdm_load_2d(unsigned lds_off, const void* gptr,
                                                   unsigned tile_w, unsigned tile_h,
                                                   unsigned long long stride_el) {
    unsigned long long ga = (unsigned long long)gptr;
    u32x4 g0;
    g0[0] = 1u;                                            // count=1, user mode
    g0[1] = lds_off;                                       // lds_addr (bytes)
    g0[2] = (unsigned)ga;                                  // global_addr[31:0]
    g0[3] = (unsigned)((ga >> 32) & 0x1FFFFFFu) | (2u << 30);  // addr[56:32] | type=2
    unsigned long long td0 = stride_el;                    // tensor_dim0 (elements)
    unsigned td1 = 1u << 20;                               // tensor_dim1 (large, no OOB)
    i32x8 g1;
    g1[0] = (int)(1u << 16);                               // data_size=1 -> 2 bytes
    g1[1] = (int)((td0 & 0xFFFFu) << 16);                  // tensor_dim0[15:0]
    g1[2] = (int)(((td0 >> 16) & 0xFFFFu) | ((td1 & 0xFFFFu) << 16));
    g1[3] = (int)(((td1 >> 16) & 0xFFFFu) | (tile_w << 16));   // tile_dim0
    g1[4] = (int)(tile_h & 0xFFFFu);                       // tile_dim1 (tile_dim2=0)
    g1[5] = (int)(unsigned)(stride_el & 0xFFFFFFFFu);      // dim0_stride[31:0]
    g1[6] = (int)((stride_el >> 32) & 0xFFFFu);            // dim0_stride[47:32]
    g1[7] = 0;
    i32x4 g2 = {0, 0, 0, 0};
    i32x4 g3 = {0, 0, 0, 0};
    i32x8 g4 = {0, 0, 0, 0, 0, 0, 0, 0};
    __builtin_amdgcn_tensor_load_to_lds(g0, g1, g2, g3, g4, 0);
}
#endif

// ---------------------------------------------------------------------------
// RMSNorm (channel dim = 128), optional cyclic shift folded into the read.
// ---------------------------------------------------------------------------
__global__ __launch_bounds__(256) void rmsnorm_k(const float* __restrict__ x,
                                                 const float* __restrict__ s,
                                                 const float* __restrict__ bb,
                                                 bf16* __restrict__ y, int shift) {
    int hw = blockIdx.x * 256 + threadIdx.x;
    int b  = blockIdx.y;
    int h = hw / WWID, w = hw % WWID;
    int src = ((h + shift) % HH) * WWID + ((w + shift) % WWID);
    const float* xb = x + (size_t)b * CCH * HWC;
    float acc = 0.f;
    for (int c = 0; c < CCH; ++c) {
        float v = xb[(size_t)c * HWC + src];
        acc += v * v;
    }
    float inv = 1.f / sqrtf(acc * (1.f / CCH) + 1e-6f);
    for (int c = 0; c < CCH; ++c) {
        float v = xb[(size_t)c * HWC + src];
        y[((size_t)b * CCH + c) * HWC + hw] = (bf16)(s[c] * v * inv + bb[c]);
    }
}

// ---------------------------------------------------------------------------
// Generic GEMM: Y[b, m, hw] = sum_k W[m,k] * X[b, k, hw] + bias[m]
// Tile 64(M) x 64(N), K-chunks of 32, 4 waves/block.  Activation tile staged
// by the Tensor Data Mover (wave 0 issues, s_wait_tensorcnt, then barrier);
// weight tile staged with fp32->bf16 conversion by all threads.
// ---------------------------------------------------------------------------
__global__ __launch_bounds__(128) void gemm_k(const float* __restrict__ Wt,
                                              const float* __restrict__ bias,
                                              const bf16* __restrict__ X,
                                              bf16* __restrict__ Y,
                                              int M, int K) {
    __shared__ bf16 sW[64][32 + 8];   // weights tile (M x Kc)
    __shared__ bf16 sX[32][64];       // activation tile (Kc x N) -- TDM row-major

    const int tid  = threadIdx.x;
    const int wave = tid >> 5;
    const int lane = tid & 31;
    const int n0 = blockIdx.x * 64;
    const int m0 = blockIdx.y * 64;
    const int b  = blockIdx.z;

    const bf16* Xb = X + (size_t)b * K * HWC;
    bf16*       Yb = Y + (size_t)b * M * HWC;

    v8f acc[4];
#pragma unroll
    for (int i = 0; i < 4; ++i)
#pragma unroll
        for (int j = 0; j < 8; ++j) acc[i][j] = 0.f;

    const int nk = K >> 5;
    for (int kk = 0; kk < nk; ++kk) {
        // ---- stage activation tile: TDM async DMA (preferred) or manual copy.
#if HAVE_TDM
        if (wave == 0) {
            tdm_load_2d((unsigned)(unsigned long long)(uintptr_t)&sX[0][0],
                        &Xb[(size_t)(kk * 32) * HWC + n0], 64u, 32u,
                        (unsigned long long)HWC);
        }
#else
#pragma unroll
        for (int i = tid; i < 32 * 64; i += 128) {
            int r = i >> 6, c = i & 63;
            sX[r][c] = Xb[(size_t)(kk * 32 + r) * HWC + (n0 + c)];
        }
#endif
        // ---- stage weight tile (fp32 -> bf16).
#pragma unroll
        for (int i = tid; i < 64 * 32; i += 128) {
            int r = i >> 5, c = i & 31;
            sW[r][c] = (bf16)Wt[(size_t)(m0 + r) * K + (kk * 32 + c)];
        }
        if (kk + 1 < nk) {
            // Speculative prefetch of next activation chunk (global_prefetch_b8).
            __builtin_prefetch(&Xb[(size_t)((kk + 1) * 32 + (tid & 31)) * HWC + n0], 0, 1);
        }
#if HAVE_TDM
        if (wave == 0) { WAIT_TENSOR(); }
#endif
        __syncthreads();

        // A fragment (16x32, ISA 7.12.2): two contiguous 8-half runs per lane
        // -> two ds_load_b128 + shuffle.
        const int mrow = wave * 16 + (lane & 15);
        const int kbA  = (lane < 16) ? 0 : 8;
        v8bf alo = *(const v8bf*)&sW[mrow][kbA];
        v8bf ahi = *(const v8bf*)&sW[mrow][16 + kbA];
        v16bf a = cat8(alo, ahi);

        const int kbB = (lane < 16) ? 0 : 8;
#pragma unroll
        for (int nt = 0; nt < 4; ++nt) {
            v16bf bfrag;
            const int ncol = nt * 16 + (lane & 15);
#pragma unroll
            for (int t = 0; t < 16; ++t) bfrag[t] = sX[kbB + (t < 8 ? t : t + 8)][ncol];
            acc[nt] = wmma_bf16(a, bfrag, acc[nt]);
        }
        __syncthreads();
    }

    // Epilogue: C/D layout (VGPR v, lane l) -> M = 16w + v + 8*(l>=16), N = l%16.
#pragma unroll
    for (int nt = 0; nt < 4; ++nt) {
#pragma unroll
        for (int v = 0; v < 8; ++v) {
            int m = m0 + wave * 16 + v + ((lane >> 4) << 3);
            int n = n0 + nt * 16 + (lane & 15);
            Yb[(size_t)m * HWC + n] = (bf16)(acc[nt][v] + bias[m]);
        }
    }
}

// ---------------------------------------------------------------------------
// Depthwise 3x3 (SAME), 512 channels, bf16 in/out, fp32 weights.
// ---------------------------------------------------------------------------
__global__ __launch_bounds__(256) void dwconv_k(const bf16* __restrict__ in,
                                                const float* __restrict__ w,
                                                const float* __restrict__ bias,
                                                bf16* __restrict__ out) {
    int hw = blockIdx.x * 256 + threadIdx.x;
    int c = blockIdx.y, b = blockIdx.z;
    int h = hw / WWID, x = hw % WWID;
    const bf16* inb = in + ((size_t)b * 512 + c) * HWC;
    float sum = bias[c];
#pragma unroll
    for (int dy = -1; dy <= 1; ++dy)
#pragma unroll
        for (int dx = -1; dx <= 1; ++dx) {
            int hh = h + dy, xx = x + dx;
            if (hh >= 0 && hh < HH && xx >= 0 && xx < WWID)
                sum += w[c * 9 + (dy + 1) * 3 + (dx + 1)] * (float)inb[hh * WWID + xx];
        }
    out[((size_t)b * 512 + c) * HWC + hw] = (bf16)sum;
}

// ---------------------------------------------------------------------------
// Gating: out[c] = u[c] * gelu(v[c]),  u = in[0:256), v = in[256:512) (erf).
// ---------------------------------------------------------------------------
__global__ __launch_bounds__(256) void gate_k(const bf16* __restrict__ in,
                                              bf16* __restrict__ out) {
    int hw = blockIdx.x * 256 + threadIdx.x;
    int c = blockIdx.y, b = blockIdx.z;
    float u = (float)in[((size_t)b * 512 + c) * HWC + hw];
    float v = (float)in[((size_t)b * 512 + 256 + c) * HWC + hw];
    float g = 0.5f * v * (1.f + erff(v * 0.70710678118654752f));
    out[((size_t)b * 256 + c) * HWC + hw] = (bf16)(u * g);
}

// ---------------------------------------------------------------------------
// Residual: xout = xin + coef * y[.., (h-shift)%H, (w-shift)%W]  (y bf16).
// ---------------------------------------------------------------------------
__global__ __launch_bounds__(256) void resid_k(const float* __restrict__ xin,
                                               const bf16* __restrict__ y,
                                               const float* __restrict__ coef,
                                               float* __restrict__ xout, int shift) {
    int hw = blockIdx.x * 256 + threadIdx.x;
    int c = blockIdx.y, b = blockIdx.z;
    int h = hw / WWID, w = hw % WWID;
    int src = ((h - shift + HH) % HH) * WWID + ((w - shift + WWID) % WWID);
    size_t o = ((size_t)b * CCH + c) * HWC;
    xout[o + hw] = xin[o + hw] + coef[0] * (float)y[o + src];
}

// ---------------------------------------------------------------------------
// Shifted-window attention. One block per window, one wave per head (4 waves).
// ---------------------------------------------------------------------------
__global__ __launch_bounds__(128) void attn_k(const bf16* __restrict__ qkv,
                                              const float* __restrict__ rpe_table,
                                              bf16* __restrict__ out) {
    __shared__ bf16 sS[4][TT * TT];   // 32KB: per-head 64x64 logits/probs

    const int tid = threadIdx.x, wave = tid >> 5, lane = tid & 31;
    const int head = wave;
    const int b = blockIdx.z;
    const int wh = blockIdx.x / (WWID / WSZ), ww = blockIdx.x % (WWID / WSZ);
    const float scale = 0.17677669529663687f;  // 32^-0.5

    const bf16* qb = qkv + ((size_t)b * 384 + 0   + head * 32) * HWC;
    const bf16* kb = qkv + ((size_t)b * 384 + 128 + head * 32) * HWC;
    const bf16* vb = qkv + ((size_t)b * 384 + 256 + head * 32) * HWC;

#define POS(p) (((wh) * WSZ + ((p) >> 3)) * WWID + (ww) * WSZ + ((p) & 7))

    // ---- Q (A-frag, 16x32) and K (B-frag, 32x16) fragments, gathered global.
    v16bf Qf[4], Kf[4];
    {
        const int db = (lane < 16) ? 0 : 8;
#pragma unroll
        for (int mt = 0; mt < 4; ++mt) {
            int dp = POS(mt * 16 + (lane & 15));
#pragma unroll
            for (int t = 0; t < 8; ++t) Qf[mt][t] = qb[(size_t)(db + t) * HWC + dp];
#pragma unroll
            for (int t = 0; t < 8; ++t) Qf[mt][8 + t] = qb[(size_t)(16 + db + t) * HWC + dp];
        }
        const int kb0 = (lane < 16) ? 0 : 8;
#pragma unroll
        for (int nt = 0; nt < 4; ++nt) {
            int dp = POS(nt * 16 + (lane & 15));
#pragma unroll
            for (int t = 0; t < 16; ++t) Kf[nt][t] = kb[(size_t)(kb0 + (t < 8 ? t : t + 8)) * HWC + dp];
        }
    }

    // ---- S = Q K^T * scale -> LDS (bf16)
#pragma unroll
    for (int mt = 0; mt < 4; ++mt) {
#pragma unroll
        for (int nt = 0; nt < 4; ++nt) {
            v8f accS;
#pragma unroll
            for (int j = 0; j < 8; ++j) accS[j] = 0.f;
            accS = wmma_bf16(Qf[mt], Kf[nt], accS);
#pragma unroll
            for (int v = 0; v < 8; ++v) {
                int qi = mt * 16 + v + ((lane >> 4) << 3);
                int ki = nt * 16 + (lane & 15);
                sS[head][qi * TT + ki] = (bf16)(accS[v] * scale);
            }
        }
    }

    // ---- softmax rows (each lane: rows lane and lane+32), RPE + shift mask.
    auto regio = [](int v) { return v < HH - WSZ ? 0 : (v < HH - SSH ? 1 : 2); };
#pragma unroll
    for (int rr = 0; rr < 2; ++rr) {
        int q = lane + rr * 32;
        int qi = q >> 3, qj = q & 7;
        int rq = regio(wh * WSZ + qi) * 3 + regio(ww * WSZ + qj);
        float mx = -1e30f;
        for (int k = 0; k < TT; ++k) {
            int ki = k >> 3, kj = k & 7;
            int rk = regio(wh * WSZ + ki) * 3 + regio(ww * WSZ + kj);
            float rpe = rpe_table[((qi - ki + 7) * 15 + (qj - kj + 7)) * 4 + head];
            float val = (float)sS[head][q * TT + k] + rpe + ((rq != rk) ? -1e30f : 0.f);
            mx = fmaxf(mx, val);
        }
        float sum = 0.f;
        for (int k = 0; k < TT; ++k) {
            int ki = k >> 3, kj = k & 7;
            int rk = regio(wh * WSZ + ki) * 3 + regio(ww * WSZ + kj);
            float rpe = rpe_table[((qi - ki + 7) * 15 + (qj - kj + 7)) * 4 + head];
            float val = (float)sS[head][q * TT + k] + rpe + ((rq != rk) ? -1e30f : 0.f);
            float e = __expf(val - mx);
            sum += e;
            sS[head][q * TT + k] = (bf16)e;   // unnormalized prob, in place
        }
        float inv = 1.f / sum;
        for (int k = 0; k < TT; ++k)
            sS[head][q * TT + k] = (bf16)((float)sS[head][q * TT + k] * inv);
    }

    // ---- V fragments (B, 32x16) for the two K-chunks and two d-tiles.
    v16bf Vf[2][2];
    {
        const int kb0 = (lane < 16) ? 0 : 8;
#pragma unroll
        for (int kc = 0; kc < 2; ++kc)
#pragma unroll
            for (int nt = 0; nt < 2; ++nt) {
                int d = nt * 16 + (lane & 15);
#pragma unroll
                for (int t = 0; t < 16; ++t)
                    Vf[kc][nt][t] = vb[(size_t)d * HWC + POS(kc * 32 + kb0 + (t < 8 ? t : t + 8))];
            }
    }

    // ---- O = P x V  (P A-fragments: two contiguous 8-half runs -> b128 loads)
    const int dbP = (lane < 16) ? 0 : 8;
#pragma unroll
    for (int mt = 0; mt < 4; ++mt) {
        v8f oacc[2];
#pragma unroll
        for (int nt = 0; nt < 2; ++nt)
#pragma unroll
            for (int j = 0; j < 8; ++j) oacc[nt][j] = 0.f;
#pragma unroll
        for (int kc = 0; kc < 2; ++kc) {
            int q = mt * 16 + (lane & 15);
            int kbase = kc * 32 + dbP;
            v8bf plo = *(const v8bf*)&sS[head][q * TT + kbase];
            v8bf phi = *(const v8bf*)&sS[head][q * TT + kbase + 16];
            v16bf pf = cat8(plo, phi);
            oacc[0] = wmma_bf16(pf, Vf[kc][0], oacc[0]);
            oacc[1] = wmma_bf16(pf, Vf[kc][1], oacc[1]);
        }
#pragma unroll
        for (int nt = 0; nt < 2; ++nt)
#pragma unroll
            for (int v = 0; v < 8; ++v) {
                int q = mt * 16 + v + ((lane >> 4) << 3);
                int d = nt * 16 + (lane & 15);
                out[((size_t)b * CCH + head * 32 + d) * HWC + POS(q)] = (bf16)oacc[nt][v];
            }
    }
#undef POS
}

// ---------------------------------------------------------------------------
// Host-side orchestration.
// ---------------------------------------------------------------------------
extern "C" void kernel_launch(void* const* d_in, const int* in_sizes, int n_in,
                              void* d_out, int out_size, void* d_ws, size_t ws_size,
                              hipStream_t stream) {
    const float* x      = (const float*)d_in[0];
    const float* cg_s   = (const float*)d_in[1];
    const float* cg_b   = (const float*)d_in[2];
    const float* pw1_w  = (const float*)d_in[3];
    const float* pw1_b  = (const float*)d_in[4];
    const float* dw_w   = (const float*)d_in[5];
    const float* dw_b   = (const float*)d_in[6];
    const float* pw2_w  = (const float*)d_in[7];
    const float* pw2_b  = (const float*)d_in[8];
    const float* beta   = (const float*)d_in[9];
    const float* at_s   = (const float*)d_in[10];
    const float* at_b   = (const float*)d_in[11];
    const float* qkv_w  = (const float*)d_in[12];
    const float* qkv_b  = (const float*)d_in[13];
    const float* rpe    = (const float*)d_in[14];
    const float* proj_w = (const float*)d_in[15];
    const float* proj_b = (const float*)d_in[16];
    const float* alpha  = (const float*)d_in[17];
    const float* ff_s   = (const float*)d_in[18];
    const float* ff_b   = (const float*)d_in[19];
    const float* fc1_w  = (const float*)d_in[20];
    const float* fc1_b  = (const float*)d_in[21];
    const float* fc2_w  = (const float*)d_in[22];
    const float* fc2_b  = (const float*)d_in[23];
    const float* gamma  = (const float*)d_in[24];

    const size_t HWc = HWC;
    bf16*  nrm   = (bf16*)d_ws;
    bf16*  g1    = nrm   + (size_t)BATCH * 128 * HWc;
    bf16*  g2    = g1    + (size_t)BATCH * 512 * HWc;
    bf16*  hid   = g2    + (size_t)BATCH * 512 * HWc;
    bf16*  yb    = hid   + (size_t)BATCH * 256 * HWc;
    bf16*  qkvb  = yb    + (size_t)BATCH * 128 * HWc;
    bf16*  attn  = qkvb  + (size_t)BATCH * 384 * HWc;
    bf16*  projo = attn  + (size_t)BATCH * 128 * HWc;
    float* x1    = (float*)(projo + (size_t)BATCH * 128 * HWc);
    float* x2    = x1 + (size_t)BATCH * 128 * HWc;
    float* outp  = (float*)d_out;

    dim3 gemm_t(128), el_t(256);

    // ---- conv_gated_block
    rmsnorm_k<<<dim3(HWC / 256, BATCH), el_t, 0, stream>>>(x, cg_s, cg_b, nrm, 0);
    gemm_k<<<dim3(HWC / 64, 512 / 64, BATCH), gemm_t, 0, stream>>>(pw1_w, pw1_b, nrm, g1, 512, 128);
    dwconv_k<<<dim3(HWC / 256, 512, BATCH), el_t, 0, stream>>>(g1, dw_w, dw_b, g2);
    gate_k<<<dim3(HWC / 256, 256, BATCH), el_t, 0, stream>>>(g2, hid);
    gemm_k<<<dim3(HWC / 64, 128 / 64, BATCH), gemm_t, 0, stream>>>(pw2_w, pw2_b, hid, yb, 128, 256);
    resid_k<<<dim3(HWC / 256, 128, BATCH), el_t, 0, stream>>>(x, yb, beta, x1, 0);

    // ---- window attention (shift folded into norm read / residual write)
    rmsnorm_k<<<dim3(HWC / 256, BATCH), el_t, 0, stream>>>(x1, at_s, at_b, nrm, SSH);
    gemm_k<<<dim3(HWC / 64, 384 / 64, BATCH), gemm_t, 0, stream>>>(qkv_w, qkv_b, nrm, qkvb, 384, 128);
    attn_k<<<dim3(NWIN, 1, BATCH), dim3(128), 0, stream>>>(qkvb, rpe, attn);
    gemm_k<<<dim3(HWC / 64, 128 / 64, BATCH), gemm_t, 0, stream>>>(proj_w, proj_b, attn, projo, 128, 128);
    resid_k<<<dim3(HWC / 256, 128, BATCH), el_t, 0, stream>>>(x1, projo, alpha, x2, SSH);

    // ---- gated FFN
    rmsnorm_k<<<dim3(HWC / 256, BATCH), el_t, 0, stream>>>(x2, ff_s, ff_b, nrm, 0);
    gemm_k<<<dim3(HWC / 64, 512 / 64, BATCH), gemm_t, 0, stream>>>(fc1_w, fc1_b, nrm, g1, 512, 128);
    gate_k<<<dim3(HWC / 256, 256, BATCH), el_t, 0, stream>>>(g1, hid);
    gemm_k<<<dim3(HWC / 64, 128 / 64, BATCH), gemm_t, 0, stream>>>(fc2_w, fc2_b, hid, yb, 128, 256);
    resid_k<<<dim3(HWC / 256, 128, BATCH), el_t, 0, stream>>>(x2, yb, gamma, outp, 0);

    (void)in_sizes; (void)n_in; (void)out_size; (void)ws_size;
}
